// Net_81939386073624
// MI455X (gfx1250) — 16-bit and 8-bit matrix operand order repaired and checked by reading, compiled, not run on hardware
//
#include <hip/hip_runtime.h>

typedef unsigned short u16;
typedef unsigned char  u8;
typedef unsigned int   u32;
typedef unsigned long long u64;

typedef __attribute__((ext_vector_type(8)))  float  v8f;
typedef __attribute__((ext_vector_type(8)))  int    v8i;
typedef __attribute__((ext_vector_type(4)))  int    v4i;
typedef __attribute__((ext_vector_type(4)))  unsigned int v4u;
typedef __attribute__((ext_vector_type(16))) __bf16 v16bf;

#if __has_builtin(__builtin_amdgcn_tensor_load_to_lds)
#define HAVE_TDM 1
#else
#define HAVE_TDM 0
#endif

#if __has_builtin(__builtin_amdgcn_rcpf)
__device__ inline float fast_rcp(float x) { return __builtin_amdgcn_rcpf(x); }
#else
__device__ inline float fast_rcp(float x) { return 1.0f / x; }
#endif

__device__ inline u16 f32_to_bf16(float x) {
  u32 u = __float_as_uint(x);
  u32 r = u + 0x7FFFu + ((u >> 16) & 1u);
  return (u16)(r >> 16);
}

__device__ inline u8 f32_to_e4m3(float x) {
  u32 u = __float_as_uint(x);
  u32 sign = (u >> 24) & 0x80u;
  u32 a = u & 0x7FFFFFFFu;
  float ax = __uint_as_float(a);
  if (ax >= 448.0f) return (u8)(sign | 0x7Eu);
  if (ax < 0.015625f) {
    int m = (int)rintf(ax * 512.0f);
    return (u8)(sign | (u32)m);
  }
  u32 r = a + 0x0007FFFFu + ((a >> 20) & 1u);
  int e = (int)(r >> 23) - 127;
  u32 mant = (r >> 20) & 7u;
  if (e > 8 || (e == 8 && mant == 7u)) return (u8)(sign | 0x7Eu);
  return (u8)(sign | ((u32)(e + 7) << 3) | mant);
}

__device__ inline float fast_sigmoid(float x) {
  return fast_rcp(1.0f + __expf(-x));
}
__device__ inline float fast_tanh(float x) {
  x = fminf(fmaxf(x, -15.0f), 15.0f);
  float e = __expf(2.0f * x);
  return 1.0f - 2.0f * fast_rcp(e + 1.0f);
}

__global__ void cvt_bf16_pad_kernel(const float* __restrict__ src, u16* __restrict__ dst,
                                    int N, int K, int Kp) {
  int idx = blockIdx.x * 256 + threadIdx.x;
  if (idx >= N * Kp) return;
  int n = idx / Kp, k = idx - n * Kp;
  dst[idx] = (k < K) ? f32_to_bf16(src[(size_t)n * K + k]) : (u16)0;
}

__global__ void cvt_fp8_kernel(const float* __restrict__ src, u8* __restrict__ dst, int n) {
  int idx = blockIdx.x * 256 + threadIdx.x;
  if (idx < n) dst[idx] = f32_to_e4m3(src[idx]);
}

__global__ void embed_kernel(const int* __restrict__ x, const float* __restrict__ emb,
                             u16* __restrict__ x0) {
  int idx = blockIdx.x * 256 + threadIdx.x;
  if (idx >= 4096 * 224) return;
  int m = idx / 224, k = idx - m * 224;
  int b = m >> 8, t = m & 255;
  float v = 0.0f;
  if (k < 200) v = emb[(size_t)x[m] * 200 + k];
  x0[(size_t)(t * 16 + b) * 224 + k] = f32_to_bf16(v);
}

template <int KP, int MODE>
__global__ __launch_bounds__(32) void gemm_bf16_kernel(
    const u16* __restrict__ A, const u16* __restrict__ W,
    const float* __restrict__ bias, float* __restrict__ C, int N) {
  int lane = threadIdx.x & 31;
  int nl = lane & 15, half = lane >> 4;
  int m0 = blockIdx.y << 4;
  int n0 = blockIdx.x << 6;

  const u16* arow = A + (size_t)(m0 + nl) * KP;
  const u16* wrow0 = W + (size_t)(n0 + nl) * KP;
  const u16* wrow1 = wrow0 + (size_t)16 * KP;
  const u16* wrow2 = wrow0 + (size_t)32 * KP;
  const u16* wrow3 = wrow0 + (size_t)48 * KP;
  int abase = half * 16;
  int bbase = half * 16;

  union FB { uint4 u[2]; v16bf v; };
  union FV { float4 f[2]; v8f v; };

  v8f acc0 = (v8f){}, acc1 = (v8f){}, acc2 = (v8f){}, acc3 = (v8f){};

#pragma unroll
  for (int kb = 0; kb < KP; kb += 32) {
    FB af, b0, b1, b2, b3;
    af.u[0] = *(const uint4*)(arow + kb + abase);
    af.u[1] = *(const uint4*)(arow + kb + abase + 8);
    b0.u[0] = *(const uint4*)(wrow0 + kb + bbase);
    b0.u[1] = *(const uint4*)(wrow0 + kb + bbase + 8);
    b1.u[0] = *(const uint4*)(wrow1 + kb + bbase);
    b1.u[1] = *(const uint4*)(wrow1 + kb + bbase + 8);
    b2.u[0] = *(const uint4*)(wrow2 + kb + bbase);
    b2.u[1] = *(const uint4*)(wrow2 + kb + bbase + 8);
    b3.u[0] = *(const uint4*)(wrow3 + kb + bbase);
    b3.u[1] = *(const uint4*)(wrow3 + kb + bbase + 8);
    acc0 = __builtin_amdgcn_wmma_f32_16x16x32_bf16(false, af.v, false, b0.v, (short)0, acc0, false, false);
    acc1 = __builtin_amdgcn_wmma_f32_16x16x32_bf16(false, af.v, false, b1.v, (short)0, acc1, false, false);
    acc2 = __builtin_amdgcn_wmma_f32_16x16x32_bf16(false, af.v, false, b2.v, (short)0, acc2, false, false);
    acc3 = __builtin_amdgcn_wmma_f32_16x16x32_bf16(false, af.v, false, b3.v, (short)0, acc3, false, false);
  }

  v8f accs[4] = {acc0, acc1, acc2, acc3};
#pragma unroll
  for (int j = 0; j < 4; ++j) {
    int n = n0 + j * 16 + nl;
    float bv = bias ? bias[n] : 0.0f;
    if (MODE == 1) {
      FV s;
#pragma unroll
      for (int r = 0; r < 8; ++r) s.v[r] = accs[j][r] + bv;
      float* base = C + (size_t)blockIdx.y * 16384 +
                    ((size_t)((n0 >> 4) + j) * 32 + lane) * 8;
      *(float4*)(base) = s.f[0];
      *(float4*)(base + 4) = s.f[1];
    } else {
#pragma unroll
      for (int r = 0; r < 8; ++r) {
        int m = m0 + r + half * 8;
        int b = m & 15, t = m >> 4;
        C[(size_t)(b * 256 + t) * N + n] = accs[j][r] + bv;
      }
    }
  }
}

#if HAVE_TDM
__device__ inline void tdm_load_xg(const float* gsrc, u32 lds_off) {
  u64 ga = (u64)(uintptr_t)gsrc;
  v4u g0;
  g0[0] = 1u;
  g0[1] = lds_off;
  g0[2] = (u32)ga;
  g0[3] = (u32)((ga >> 32) & 0x1FFFFFFu) | (2u << 30);
  v8i g1;
  g1[0] = (int)(2u << 16);
  g1[1] = (int)(0x4000u << 16);
  g1[2] = (int)(1u << 16);
  g1[3] = (int)(0x4000u << 16);
  g1[4] = 1;
  g1[5] = 16384;
  g1[6] = 0;
  g1[7] = 0;
  v4i z4 = (v4i){0, 0, 0, 0};
#if __has_include(<hip/amd_detail/amd_gfx1250_TDM.h>)
  v8i z8 = (v8i){0, 0, 0, 0, 0, 0, 0, 0};
  __builtin_amdgcn_tensor_load_to_lds(g0, g1, z4, z4, z8, 0);
#else
  __builtin_amdgcn_tensor_load_to_lds(g0, g1, z4, z4, 0);
#endif
}
#endif

__global__ __launch_bounds__(512, 1) void lstm_seq_kernel(
    const float* __restrict__ xg, const u8* __restrict__ wh8,
    u16* __restrict__ xout) {
  __shared__ __align__(16) float xgbuf[2][16 * 1024];
  __shared__ __align__(16) u8 h8[16 * 272];
  int tid = threadIdx.x;
  int wave = tid >> 5, lane = tid & 31;
  int nl = lane & 15, half = lane >> 4;

  union FI { uint4 u[2]; u64 d[4]; v8i v; };
  union FV { float4 f[2]; v8f v; };

  v8i Bf[4][4];
#pragma unroll
  for (int g = 0; g < 4; ++g) {
    const u8* wrow = wh8 + (size_t)(g * 256 + wave * 16 + nl) * 256;
#pragma unroll
    for (int kc = 0; kc < 4; ++kc) {
      FI t;
      t.u[0] = *(const uint4*)(wrow + kc * 64 + half * 16);
      t.u[1] = *(const uint4*)(wrow + kc * 64 + 32 + half * 16);
      Bf[g][kc] = t.v;
    }
  }

#if HAVE_TDM
  if (wave == 0)
    tdm_load_xg(xg, (u32)(uintptr_t)&xgbuf[0][0]);
#endif

  for (int i = tid; i < 16 * 272; i += 512) h8[i] = 0;

  v8f cst = (v8f){};
  int j = wave * 16 + nl;
  int arow = nl * 272;
  int ab = half * 8;

  for (int t = 0; t < 256; ++t) {
#if HAVE_TDM
    if (wave == 0) __builtin_amdgcn_s_wait_tensorcnt(0);
#endif
    __syncthreads();
#if HAVE_TDM
    if (wave == 0 && t + 1 < 256)
      tdm_load_xg(xg + (size_t)(t + 1) * 16 * 1024,
                  (u32)(uintptr_t)&xgbuf[(t + 1) & 1][0]);
    const float* xgt = &xgbuf[t & 1][0];
#else
    const float* xgt = xg + (size_t)t * 16 * 1024;
#endif

    v8f acc[4];
#pragma unroll
    for (int g = 0; g < 4; ++g) {
      const float* p = xgt + ((size_t)(g * 16 + wave) * 32 + lane) * 8;
      FV s;
      s.f[0] = *(const float4*)(p);
      s.f[1] = *(const float4*)(p + 4);
      acc[g] = s.v;
    }

    v8i Af[4];
#pragma unroll
    for (int kc = 0; kc < 4; ++kc) {
      FI a;
#pragma unroll
      for (int grp = 0; grp < 4; ++grp)
        a.d[grp] = *(const u64*)(&h8[arow + kc * 64 + (grp >> 1) * 32 + (grp & 1) * 8 + 2 * ab]);
      Af[kc] = a.v;
    }

#pragma unroll
    for (int kc = 0; kc < 4; ++kc)
#pragma unroll
      for (int g = 0; g < 4; ++g)
        acc[g] = __builtin_amdgcn_wmma_f32_16x16x64_fp8_fp8(
            Af[kc], Bf[g][kc], (short)0, acc[g], false, false);

    __syncthreads();

#pragma unroll
    for (int r = 0; r < 8; ++r) {
      int b = r + half * 8;
      float ig = fast_sigmoid(acc[0][r]);
      float fg = fast_sigmoid(acc[1][r]);
      float gg = fast_tanh(acc[2][r]);
      float og = fast_sigmoid(acc[3][r]);
      float c = fg * cst[r] + ig * gg;
      cst[r] = c;
      float h = og * fast_tanh(c);
      h8[b * 272 + j] = f32_to_e4m3(h);
      xout[((size_t)t * 16 + b) * 256 + j] = f32_to_bf16(h);
    }
  }
}

__global__ __launch_bounds__(256) void logsoftmax_kernel(float* __restrict__ out) {
  const int V = 32000;
  float* p = out + (size_t)blockIdx.x * V;
  __shared__ float red[8];
  __shared__ float bval;
  int tid = threadIdx.x, lane = tid & 31, w = tid >> 5;

  float mx = -3.0e38f;
  for (int c = tid; c < V; c += 256) mx = fmaxf(mx, p[c]);
#pragma unroll
  for (int s = 16; s > 0; s >>= 1) mx = fmaxf(mx, __shfl_xor(mx, s, 32));
  if (lane == 0) red[w] = mx;
  __syncthreads();
  if (tid == 0) {
    float m = red[0];
    for (int i = 1; i < 8; ++i) m = fmaxf(m, red[i]);
    bval = m;
  }
  __syncthreads();
  float rowmax = bval;

  float sum = 0.0f;
  for (int c = tid; c < V; c += 256) sum += __expf(p[c] - rowmax);
#pragma unroll
  for (int s = 16; s > 0; s >>= 1) sum += __shfl_xor(sum, s, 32);
  if (lane == 0) red[w] = sum;
  __syncthreads();
  if (tid == 0) {
    float s = 0.0f;
    for (int i = 0; i < 8; ++i) s += red[i];
    bval = logf(s);
  }
  __syncthreads();
  float lse = bval;
  for (int c = tid; c < V; c += 256) p[c] = p[c] - rowmax - lse;
}

extern "C" void kernel_launch(void* const* d_in, const int* in_sizes, int n_in,
                              void* d_out, int out_size, void* d_ws, size_t ws_size,
                              hipStream_t stream) {
  (void)in_sizes; (void)n_in; (void)out_size; (void)ws_size;
  const int*   x    = (const int*)d_in[0];
  const float* emb  = (const float*)d_in[1];
  const float* Wi[3] = {(const float*)d_in[2], (const float*)d_in[5], (const float*)d_in[8]};
  const float* Wh[3] = {(const float*)d_in[3], (const float*)d_in[6], (const float*)d_in[9]};
  const float* bb[3] = {(const float*)d_in[4], (const float*)d_in[7], (const float*)d_in[10]};
  const float* fcW  = (const float*)d_in[11];
  const float* fcb  = (const float*)d_in[12];
  float* out = (float*)d_out;

  char* ws = (char*)d_ws;
  size_t off = 0;
  auto carve = [&](size_t bytes) -> void* {
    void* p = ws + off; off += (bytes + 255) & ~(size_t)255; return p;
  };
  u16*  xA    = (u16*)carve((size_t)4096 * 256 * 2);
  u16*  xB    = (u16*)carve((size_t)4096 * 256 * 2);
  float* xg   = (float*)carve((size_t)4096 * 1024 * 4);
  u16*  wibf  = (u16*)carve((size_t)1024 * 256 * 2);
  u8*   wh8   = (u8*)carve((size_t)1024 * 256);
  u16*  fcwbf = (u16*)carve((size_t)32000 * 256 * 2);

  cvt_bf16_pad_kernel<<<(32000 * 256) / 256, 256, 0, stream>>>(fcW, fcwbf, 32000, 256, 256);
  embed_kernel<<<(4096 * 224) / 256, 256, 0, stream>>>(x, emb, xA);

  u16* cur = xA; u16* nxt = xB;
  for (int l = 0; l < 3; ++l) {
    const int Kp = (l == 0) ? 224 : 256;
    const int Ksrc = (l == 0) ? 200 : 256;
    cvt_bf16_pad_kernel<<<(1024 * Kp) / 256, 256, 0, stream>>>(Wi[l], wibf, 1024, Ksrc, Kp);
    if (l == 0)
      gemm_bf16_kernel<224, 1><<<dim3(1024 / 64, 4096 / 16), 32, 0, stream>>>(
          cur, wibf, bb[l], xg, 1024);
    else
      gemm_bf16_kernel<256, 1><<<dim3(1024 / 64, 4096 / 16), 32, 0, stream>>>(
          cur, wibf, bb[l], xg, 1024);
    cvt_fp8_kernel<<<(1024 * 256) / 256, 256, 0, stream>>>(Wh[l], wh8, 1024 * 256);
    lstm_seq_kernel<<<1, 512, 0, stream>>>(xg, wh8, nxt);
    u16* tmp = cur; cur = nxt; nxt = tmp;
  }

  gemm_bf16_kernel<256, 0><<<dim3(32000 / 64, 4096 / 16), 32, 0, stream>>>(
      cur, fcwbf, fcb, out, 32000);
  logsoftmax_kernel<<<4096, 256, 0, stream>>>(out);
}
